// SelfAttention_43344809951876
// MI455X (gfx1250) — matrix operand order repaired and checked by reading, not compiled, not run on hardware
//
#include <hip/hip_runtime.h>

#define B_    2
#define N_    2048
#define DIM_  1024
#define H_    16
#define HD_   64
#define SCALE_ 0.125f   // 64^-0.5

typedef __attribute__((ext_vector_type(16))) __bf16 v16bf;
typedef __attribute__((ext_vector_type(8)))  float  v8f;

union BF16Frag { v16bf v; unsigned int u[8]; };

__device__ __forceinline__ unsigned short f2bf(float f) {
  unsigned int u = __float_as_uint(f);
  u += 0x7FFFu + ((u >> 16) & 1u);   // round-to-nearest-even
  return (unsigned short)(u >> 16);
}

__device__ __forceinline__ v8f wmma_bf16(const BF16Frag& a, const BF16Frag& b, v8f c) {
  return __builtin_amdgcn_wmma_f32_16x16x32_bf16(false, a.v, false, b.v,
                                                 (short)0, c, false, false);
}

// ---------------- conversion kernels ----------------

__global__ __launch_bounds__(256) void cvt_f32_bf16(const float* __restrict__ src,
                                                    unsigned short* __restrict__ dst,
                                                    int n) {
  int i = blockIdx.x * 256 + threadIdx.x;
  if (i < n) dst[i] = f2bf(src[i]);
}

// w is [in_dim, out_dim] row-major (f32); wT is [out_dim, in_dim] bf16
__global__ __launch_bounds__(256) void cvt_transpose_bf16(const float* __restrict__ w,
                                                          unsigned short* __restrict__ wT,
                                                          int in_dim, int out_dim) {
  int idx = blockIdx.x * 256 + threadIdx.x;
  if (idx < in_dim * out_dim) {
    int o = idx / in_dim;
    int i = idx - o * in_dim;
    wT[idx] = f2bf(w[(size_t)i * out_dim + o]);
  }
}

// ---------------- async-staged 64x128 block-tile bf16 GEMM core ----------------
// A: [M,K] bf16 row-major; Bt: [Ncols,K] bf16 row-major (B transposed).
// 128 threads = 4 waves; wave w -> rows 32*(w/2), cols 64*(w%2);
// per wave: 2 A-frags x 4 B-frags -> 8 wmma accumulators per K-step.
// LDS: 2 buffers x (A 64x32 + B 128x32), rows padded to 80B (20 u32) ->
//      bank-conflict-free fragment reads, 16B-aligned async B128 writes.

#define LDS_ROW_U32 20              // 64B data + 16B pad
#define LDS_A_BYTES (64 * 80)       // 5120
#define LDS_B_BYTES (128 * 80)      // 10240
#define LDS_BUF_BYTES (LDS_A_BYTES + LDS_B_BYTES)   // 15360
#define LDS_BUF_U32 (LDS_BUF_BYTES / 4)             // 3840

__device__ __forceinline__ void issue_tile_async(const unsigned short* __restrict__ gA,
                                                 const unsigned short* __restrict__ gB,
                                                 int K, int rowBase, int colBase, int k0,
                                                 unsigned ldsDstA, unsigned ldsDstB,
                                                 int tid) {
  // A: 256 B128 transfers, B: 512; 128 threads -> 6 async instr per wave.
#pragma unroll
  for (int it = 0; it < 2; ++it) {
    int ti  = tid + it * 128;
    int row = ti >> 2;            // 0..63
    int ch  = ti & 3;             // 16B chunk within 64B row
    unsigned long long ga = (unsigned long long)(uintptr_t)gA +
        (((unsigned long long)(rowBase + row) * K + k0) << 1) + (unsigned)(ch << 4);
    unsigned la = ldsDstA + (unsigned)(row * 80 + (ch << 4));
    asm volatile("global_load_async_to_lds_b128 %0, %1, off"
                 :: "v"(la), "v"(ga) : "memory");
  }
#pragma unroll
  for (int it = 0; it < 4; ++it) {
    int ti  = tid + it * 128;
    int row = ti >> 2;            // 0..127
    int ch  = ti & 3;
    unsigned long long gb = (unsigned long long)(uintptr_t)gB +
        (((unsigned long long)(colBase + row) * K + k0) << 1) + (unsigned)(ch << 4);
    unsigned lb = ldsDstB + (unsigned)(row * 80 + (ch << 4));
    asm volatile("global_load_async_to_lds_b128 %0, %1, off"
                 :: "v"(lb), "v"(gb) : "memory");
  }
}

__device__ __forceinline__ void gemm_step(const unsigned int* __restrict__ sAB, int cur,
                                          int rg, int cg, int hf, int mn,
                                          v8f acc[2][4]) {
  const unsigned int* sAu = sAB + cur * LDS_BUF_U32;
  const unsigned int* sBu = sAB + cur * LDS_BUF_U32 + (LDS_A_BYTES >> 2);
  BF16Frag a[2], b[4];
#pragma unroll
  for (int v = 0; v < 8; ++v) {
    // A frag: K pair base = 2v + 16*half
    int kidx = 2 * v + 16 * hf;
    a[0].u[v] = sAu[(rg * 32 +      mn) * LDS_ROW_U32 + (kidx >> 1)];
    a[1].u[v] = sAu[(rg * 32 + 16 + mn) * LDS_ROW_U32 + (kidx >> 1)];
    // B frag: K pair base = 2v + 16*half -> u32 index v + 8*half
    int bi = v + 8 * hf;
#pragma unroll
    for (int ci = 0; ci < 4; ++ci)
      b[ci].u[v] = sBu[(cg * 64 + ci * 16 + mn) * LDS_ROW_U32 + bi];
  }
#pragma unroll
  for (int ri = 0; ri < 2; ++ri)
#pragma unroll
    for (int ci = 0; ci < 4; ++ci)
      acc[ri][ci] = wmma_bf16(a[ri], b[ci], acc[ri][ci]);
}

__device__ __forceinline__ void gemm64x128_async(const unsigned short* __restrict__ A,
                                                 const unsigned short* __restrict__ Bt,
                                                 int K, int rowBase, int colBase,
                                                 unsigned int* sAB, v8f acc[2][4]) {
  const int tid  = threadIdx.x;
  const int lane = tid & 31;
  const int wv   = tid >> 5;      // 0..3
  const int hf   = lane >> 4;     // lane half
  const int mn   = lane & 15;
  const int rg   = wv >> 1;       // 32-row group
  const int cg   = wv & 1;        // 64-col group
  const unsigned ldsBase = (unsigned)(uintptr_t)sAB;   // low 32 bits = LDS offset

  // prologue: stream tile 0 into buffer 0
  issue_tile_async(A, Bt, K, rowBase, colBase, 0, ldsBase, ldsBase + LDS_A_BYTES, tid);

  int cur = 0;
  int k0 = 0;
  for (; k0 + 32 < K; k0 += 32) {        // steady state: branch-free waits
    int nxt = cur ^ 1;
    issue_tile_async(A, Bt, K, rowBase, colBase, k0 + 32,
                     ldsBase + nxt * LDS_BUF_BYTES,
                     ldsBase + nxt * LDS_BUF_BYTES + LDS_A_BYTES, tid);
    asm volatile("s_wait_asynccnt 0x6" ::: "memory");  // current tile landed
    __syncthreads();
    gemm_step(sAB, cur, rg, cg, hf, mn, acc);
    __syncthreads();   // everyone done reading buf before it is refilled
    cur = nxt;
  }
  // epilogue step
  asm volatile("s_wait_asynccnt 0x0" ::: "memory");
  __syncthreads();
  gemm_step(sAB, cur, rg, cg, hf, mn, acc);
}

// ---------------- QKV projection ----------------
// out cols: s*DIM + h*HD + hd; writes q,k as [bh][n][hd], v transposed [bh][hd][n]
__global__ __launch_bounds__(128) void qkv_gemm_kernel(const unsigned short* __restrict__ xb,
                                                       const unsigned short* __restrict__ wT,
                                                       unsigned short* __restrict__ qb,
                                                       unsigned short* __restrict__ kbuf,
                                                       unsigned short* __restrict__ vT) {
  __shared__ unsigned int sAB[2 * LDS_BUF_U32];
  v8f acc[2][4];
#pragma unroll
  for (int ri = 0; ri < 2; ++ri)
#pragma unroll
    for (int ci = 0; ci < 4; ++ci) acc[ri][ci] = (v8f){0, 0, 0, 0, 0, 0, 0, 0};
  const int rowBase = blockIdx.x * 64;
  const int colBase = blockIdx.y * 128;
  gemm64x128_async(xb, wT, DIM_, rowBase, colBase, sAB, acc);

  const int lane = threadIdx.x & 31;
  const int wv   = threadIdx.x >> 5;
  const int hf   = lane >> 4;
  const int mn   = lane & 15;
  const int rg   = wv >> 1, cg = wv & 1;
#pragma unroll
  for (int ri = 0; ri < 2; ++ri) {
#pragma unroll
    for (int ci = 0; ci < 4; ++ci) {
      v8f c = acc[ri][ci];
      int gcol = colBase + cg * 64 + ci * 16 + mn;
      int s    = gcol / DIM_;
      int rem  = gcol - s * DIM_;
      int h    = rem >> 6;      // / HD
      int hd   = rem & 63;
#pragma unroll
      for (int r = 0; r < 8; ++r) {
        int grow = rowBase + rg * 32 + ri * 16 + r + 8 * hf;
        int bb = grow >> 11;    // / N
        int nn = grow & 2047;
        size_t bh = (size_t)bb * H_ + h;
        unsigned short val = f2bf(c[r]);
        if (s == 0)      qb  [(bh * N_ + nn) * HD_ + hd] = val;
        else if (s == 1) kbuf[(bh * N_ + nn) * HD_ + hd] = val;
        else             vT  [(bh * HD_ + hd) * N_ + nn] = val;
      }
    }
  }
}

// ---------------- flash attention ----------------
// one wave per (b,h,16-row tile); 128 thr/block = 4 waves; per-wave LDS P stage
__global__ __launch_bounds__(128) void attn_kernel(const unsigned short* __restrict__ Q,
                                                   const unsigned short* __restrict__ Kt,
                                                   const unsigned short* __restrict__ Vt,
                                                   unsigned short* __restrict__ ctx) {
  __shared__ unsigned int Ps[4 * 256];   // 4 waves x (16x32 bf16)
  const int tid  = threadIdx.x;
  const int lane = tid & 31;
  const int wib  = tid >> 5;
  const int hf   = lane >> 4;
  const int mn   = lane & 15;
  const int wid  = blockIdx.x * 4 + wib;
  const int bh   = wid >> 7;     // / (N/16)
  const int tile = wid & 127;
  const int rowBase = tile << 4;
  unsigned int* myP = Ps + wib * 256;

  // load Q tile 16x64 as two A-fragments (K = hd 0..31 / 32..63)
  BF16Frag qa0, qa1;
  const unsigned int* qg = (const unsigned int*)(Q + ((size_t)bh * N_ + rowBase) * HD_);
#pragma unroll
  for (int v = 0; v < 8; ++v) {
    int kidx = 2 * v + 16 * hf;
    qa0.u[v] = qg[mn * 32 + (kidx >> 1)];
    qa1.u[v] = qg[mn * 32 + 16 + (kidx >> 1)];
  }

  float mrow[8], lrow[8], alpha[8];
#pragma unroll
  for (int r = 0; r < 8; ++r) { mrow[r] = -1e30f; lrow[r] = 0.f; }
  v8f o0 = {0, 0, 0, 0, 0, 0, 0, 0}, o1 = o0, o2 = o0, o3 = o0;

  for (int kc = 0; kc < N_; kc += 32) {
    // scores S[16 x 32] = Q (16x64) @ K^T (64x32)
    v8f s0 = {0, 0, 0, 0, 0, 0, 0, 0}, s1 = s0;
#pragma unroll
    for (int t = 0; t < 2; ++t) {
      const unsigned int* kg =
          (const unsigned int*)(Kt + ((size_t)bh * N_ + kc + 16 * t) * HD_);
      BF16Frag f0, f1;
#pragma unroll
      for (int v = 0; v < 8; ++v) {
        int bi = v + 8 * hf;                    // B frag: k = hd pair 2v+16*half
        f0.u[v] = kg[mn * 32 + bi];             // hd 0..31
        f1.u[v] = kg[mn * 32 + 16 + bi];        // hd 32..63
      }
      if (t == 0) { s0 = wmma_bf16(qa0, f0, s0); s0 = wmma_bf16(qa1, f1, s0); }
      else        { s1 = wmma_bf16(qa0, f0, s1); s1 = wmma_bf16(qa1, f1, s1); }
    }

    // online softmax update (row r+8*half lives in vgpr r across 16 lanes)
#pragma unroll
    for (int r = 0; r < 8; ++r) {
      float a0 = s0[r] * SCALE_, a1 = s1[r] * SCALE_;
      float mx = fmaxf(a0, a1);
#pragma unroll
      for (int off = 1; off < 16; off <<= 1) mx = fmaxf(mx, __shfl_xor(mx, off, 32));
      float mnew = fmaxf(mrow[r], mx);
      alpha[r] = __expf(mrow[r] - mnew);
      mrow[r]  = mnew;
      float p0 = __expf(a0 - mnew), p1 = __expf(a1 - mnew);
      s0[r] = p0; s1[r] = p1;
      float sum = p0 + p1;
#pragma unroll
      for (int off = 1; off < 16; off <<= 1) sum += __shfl_xor(sum, off, 32);
      lrow[r] = lrow[r] * alpha[r] + sum;
    }
#pragma unroll
    for (int r = 0; r < 8; ++r) {
      o0[r] *= alpha[r]; o1[r] *= alpha[r]; o2[r] *= alpha[r]; o3[r] *= alpha[r];
    }

    // bounce P through LDS: D-layout -> A-fragment layout (wave-private, in-order)
    unsigned short* myP16 = (unsigned short*)myP;
#pragma unroll
    for (int r = 0; r < 8; ++r) {
      int row = r + 8 * hf;
      myP16[row * 32 + mn]      = f2bf(s0[r]);
      myP16[row * 32 + 16 + mn] = f2bf(s1[r]);
    }
    BF16Frag pa;
#pragma unroll
    for (int v = 0; v < 8; ++v) {
      int kidx = 2 * v + 16 * hf;   // k = key 0..31
      pa.u[v] = myP[mn * 16 + (kidx >> 1)];
    }

    // O[16x64] += P (16x32) @ V (32x64); V stored transposed [bh][hd][n]
#pragma unroll
    for (int d = 0; d < 4; ++d) {
      BF16Frag fb;
      const unsigned int* vg =
          (const unsigned int*)(Vt + ((size_t)bh * HD_ + d * 16 + mn) * N_ + kc);
#pragma unroll
      for (int v = 0; v < 8; ++v) fb.u[v] = vg[v + 8 * hf];  // keys 2v+16*half
      if (d == 0) o0 = wmma_bf16(pa, fb, o0);
      else if (d == 1) o1 = wmma_bf16(pa, fb, o1);
      else if (d == 2) o2 = wmma_bf16(pa, fb, o2);
      else o3 = wmma_bf16(pa, fb, o3);
    }
  }

  // epilogue: normalize and write ctx [b, n, DIM] bf16
  const int b = bh >> 4;
  const int h = bh & 15;
#pragma unroll
  for (int d = 0; d < 4; ++d) {
    v8f o = (d == 0) ? o0 : (d == 1) ? o1 : (d == 2) ? o2 : o3;
#pragma unroll
    for (int r = 0; r < 8; ++r) {
      int row = rowBase + r + 8 * hf;
      int col = h * HD_ + d * 16 + mn;
      ctx[((size_t)b * N_ + row) * DIM_ + col] = f2bf(o[r] / lrow[r]);
    }
  }
}

// ---------------- output projection + bias ----------------
__global__ __launch_bounds__(128) void out_proj_kernel(const unsigned short* __restrict__ ctx,
                                                       const unsigned short* __restrict__ woT,
                                                       const float* __restrict__ b_out,
                                                       float* __restrict__ out) {
  __shared__ unsigned int sAB[2 * LDS_BUF_U32];
  v8f acc[2][4];
#pragma unroll
  for (int ri = 0; ri < 2; ++ri)
#pragma unroll
    for (int ci = 0; ci < 4; ++ci) acc[ri][ci] = (v8f){0, 0, 0, 0, 0, 0, 0, 0};
  const int rowBase = blockIdx.x * 64;
  const int colBase = blockIdx.y * 128;
  gemm64x128_async(ctx, woT, DIM_, rowBase, colBase, sAB, acc);

  const int lane = threadIdx.x & 31;
  const int wv   = threadIdx.x >> 5;
  const int hf   = lane >> 4;
  const int mn   = lane & 15;
  const int rg   = wv >> 1, cg = wv & 1;
#pragma unroll
  for (int ri = 0; ri < 2; ++ri) {
#pragma unroll
    for (int ci = 0; ci < 4; ++ci) {
      v8f c = acc[ri][ci];
      int gcol   = colBase + cg * 64 + ci * 16 + mn;
      float bias = b_out[gcol];
#pragma unroll
      for (int r = 0; r < 8; ++r) {
        int grow = rowBase + rg * 32 + ri * 16 + r + 8 * hf;
        out[(size_t)grow * DIM_ + gcol] = c[r] + bias;
      }
    }
  }
}

// ---------------- host launch ----------------
extern "C" void kernel_launch(void* const* d_in, const int* in_sizes, int n_in,
                              void* d_out, int out_size, void* d_ws, size_t ws_size,
                              hipStream_t stream) {
  (void)in_sizes; (void)n_in; (void)out_size; (void)ws_size;
  const float* x     = (const float*)d_in[0];
  const float* w_qkv = (const float*)d_in[1];
  const float* w_out = (const float*)d_in[2];
  const float* b_out = (const float*)d_in[3];
  float* out = (float*)d_out;

  // workspace carve (bf16 elements): total 25,165,824 u16 = 48 MB
  unsigned short* ws16 = (unsigned short*)d_ws;
  const size_t RN = (size_t)B_ * N_;                 // 4096 rows
  unsigned short* xb    = ws16;                      // RN*DIM
  unsigned short* wqkvT = xb + RN * DIM_;            // 3*DIM*DIM, [out][in]
  unsigned short* woT   = wqkvT + (size_t)3 * DIM_ * DIM_;  // DIM*DIM, [out][in]
  unsigned short* qb    = woT + (size_t)DIM_ * DIM_; // [bh][n][hd]
  unsigned short* kbuf  = qb + RN * DIM_;            // [bh][n][hd]
  unsigned short* vT    = kbuf + RN * DIM_;          // [bh][hd][n]
  unsigned short* ctx   = vT + RN * DIM_;            // [b*n][DIM]

  const int nx = (int)(RN * DIM_);
  cvt_f32_bf16<<<(nx + 255) / 256, 256, 0, stream>>>(x, xb, nx);
  cvt_transpose_bf16<<<(3 * DIM_ * DIM_ + 255) / 256, 256, 0, stream>>>(
      w_qkv, wqkvT, DIM_, 3 * DIM_);
  cvt_transpose_bf16<<<(DIM_ * DIM_ + 255) / 256, 256, 0, stream>>>(
      w_out, woT, DIM_, DIM_);

  qkv_gemm_kernel<<<dim3(64, 24), 128, 0, stream>>>(xb, wqkvT, qb, kbuf, vT);
  attn_kernel<<<1024, 128, 0, stream>>>(qb, kbuf, vT, ctx);
  out_proj_kernel<<<dim3(64, 8), 128, 0, stream>>>(ctx, woT, b_out, out);
}
